// MultiHeadAttention_43250320670891
// MI455X (gfx1250) — matrix operand order repaired and checked by reading, not compiled, not run on hardware
//
#include <hip/hip_runtime.h>

// ---------------------------------------------------------------------------
// MI455X (gfx1250) multi-head attention, bf16 WMMA + async-to-LDS staging.
// wave32; WMMA shape 16x16x32 bf16; ASYNCcnt-pipelined K/V tiles; weights
// pre-converted to bf16 once so GEMM staging is pure async DMA.
// ---------------------------------------------------------------------------

typedef __attribute__((ext_vector_type(16))) __bf16 v16bf;
typedef __attribute__((ext_vector_type(2)))  __bf16 v2bf;
typedef __attribute__((ext_vector_type(8)))  float  v8f;

#define DMODEL 1024
#define HEADS  16
#define DK     64
#define BATCH  2
#define SEQ    2048
#define MROWS  (BATCH * SEQ)          // 4096 token rows

#define TSTRIDE 72                    // ushorts/row in LDS tiles (144B, 16B aligned)

union Frag16 { v16bf v; unsigned short u[16]; uint4 q[2]; };
union Acc8   { v8f  v; float f[8]; };

__device__ __forceinline__ unsigned short f2bf(float f) {
    unsigned u = __builtin_bit_cast(unsigned, f);
    u += 0x7FFFu + ((u >> 16) & 1u);          // round-to-nearest-even
    return (unsigned short)(u >> 16);
}

// Packed f32x2 -> bf16x2: one VALU op (v_cvt_pk_bf16_f32).
__device__ __forceinline__ unsigned f2bf_pk(float lo, float hi) {
#if __has_builtin(__builtin_amdgcn_cvt_pk_bf16_f32)
    v2bf p = __builtin_amdgcn_cvt_pk_bf16_f32(lo, hi);
    return __builtin_bit_cast(unsigned, p);
#else
    unsigned r;
    asm("v_cvt_pk_bf16_f32 %0, %1, %2" : "=v"(r) : "v"(lo), "v"(hi));
    return r;
#endif
}

__device__ __forceinline__ v8f wmma_bf16(v16bf a, v16bf b, v8f c) {
    return __builtin_amdgcn_wmma_f32_16x16x32_bf16(
        false, a, false, b, (short)0, c, false, false);
}

// Async DMA: 16 bytes global -> LDS, tracked by ASYNCcnt (CDNA5).
__device__ __forceinline__ void async_load_b128(const void* g, unsigned lds) {
    asm volatile("global_load_async_to_lds_b128 %0, %1, off"
                 :: "v"(lds), "v"(g) : "memory");
}
__device__ __forceinline__ unsigned lds_addr(const void* p) {
    return (unsigned)(size_t)p;       // low 32 bits = LDS byte address
}

// A-matrix fragment (16x32 bf16, MxK) from a row-major bf16 tile.
// lane<16 -> k0=0, lane>=16 -> k0=16; halves 0-15 = K k0..k0+15.
__device__ __forceinline__ v16bf load_frag_a(const unsigned short* t, int stride,
                                             int mBase, int kBase) {
    const int lane = threadIdx.x & 31;
    const int m  = mBase + (lane & 15);
    const int k0 = kBase + ((lane < 16) ? 0 : 16);
    const unsigned short* p = t + m * stride + k0;
    Frag16 f;
    f.q[0] = *(const uint4*)(p);
    f.q[1] = *(const uint4*)(p + 8);
    return f.v;
}

// B-matrix fragment (32x16 bf16, KxN) from a tile storing B^T row-major
// (row index = N column, columns = K). lane<16 -> k0=0, lane>=16 -> k0=16.
__device__ __forceinline__ v16bf load_frag_b(const unsigned short* t, int stride,
                                             int nBase, int kBase) {
    const int lane = threadIdx.x & 31;
    const int n  = nBase + (lane & 15);
    const int k0 = kBase + ((lane < 16) ? 0 : 16);
    const unsigned short* p = t + n * stride + k0;
    Frag16 f;
    f.q[0] = *(const uint4*)(p);
    f.q[1] = *(const uint4*)(p + 8);
    return f.v;
}

// B-matrix fragment from a tile storing B row-major ([K][N], e.g. V tiles):
// strided 16-bit reads down column n.
__device__ __forceinline__ v16bf load_frag_b_rowmajor(const unsigned short* t,
                                                      int stride, int nBase,
                                                      int kBase) {
    const int lane = threadIdx.x & 31;
    const int n  = nBase + (lane & 15);
    const int k0 = kBase + ((lane < 16) ? 0 : 16);
    Frag16 f;
    #pragma unroll
    for (int h = 0; h < 16; ++h)
        f.u[h] = t[(k0 + h) * stride + n];
    return f.v;
}

// ---------------------------------------------------------------------------
// Kernel 0: one-shot f32 -> bf16 conversion (weights).  Bandwidth-bound.
// ---------------------------------------------------------------------------
__global__ __launch_bounds__(256) void cvt_f32_bf16(
    const float* __restrict__ src, unsigned short* __restrict__ dst, int n)
{
    const int i = (blockIdx.x * 256 + threadIdx.x) * 8;
    if (i < n) {
        float4 a = *(const float4*)(src + i);
        float4 b = *(const float4*)(src + i + 4);
        unsigned* d = (unsigned*)(dst + i);
        d[0] = f2bf_pk(a.x, a.y);
        d[1] = f2bf_pk(a.z, a.w);
        d[2] = f2bf_pk(b.x, b.y);
        d[3] = f2bf_pk(b.z, b.w);
    }
}

// ---------------------------------------------------------------------------
// Kernel 1: QKV projections.  out = (x @ W^T + b) [* 1/sqrt(dk) for Q]
// bf16 out, head-major [B, H, S, DK].  blockIdx.z selects Q/K/V.
// Block tile 64(M) x 128(N), 4 waves; each wave = 2x4 WMMA tiles, K-step 64.
// A tile: f32 x converted in-flight (packed cvt).  B tile: bf16 W, async DMA.
// ---------------------------------------------------------------------------
__global__ __launch_bounds__(128) void mha_qkv_proj(
    const float* __restrict__ q_in, const float* __restrict__ k_in,
    const float* __restrict__ v_in,
    const unsigned short* __restrict__ Wqb, const unsigned short* __restrict__ Wkb,
    const unsigned short* __restrict__ Wvb,
    const float* __restrict__ bq, const float* __restrict__ bk,
    const float* __restrict__ bv,
    unsigned short* __restrict__ Qo, unsigned short* __restrict__ Ko,
    unsigned short* __restrict__ Vo)
{
    const int z = blockIdx.z;
    const float* x    = (z == 0) ? q_in : (z == 1) ? k_in : v_in;
    const unsigned short* W = (z == 0) ? Wqb : (z == 1) ? Wkb : Wvb;
    const float* bias = (z == 0) ? bq   : (z == 1) ? bk   : bv;
    unsigned short* out = (z == 0) ? Qo : (z == 1) ? Ko : Vo;
    const float outScale = (z == 0) ? 0.125f : 1.0f;   // 1/sqrt(64)

    __shared__ unsigned short As[64 * TSTRIDE];
    __shared__ unsigned short Bs[128 * TSTRIDE];

    const int tid   = threadIdx.x;
    const int wave  = tid >> 5;
    const int lane  = tid & 31;
    const int mBase = blockIdx.x * 64;
    const int nBase = blockIdx.y * 128;
    const int wm = (wave & 1) * 32;
    const int wn = (wave >> 1) * 64;

    v8f acc[2][4];
    const v8f vzero = {0.f,0.f,0.f,0.f,0.f,0.f,0.f,0.f};
    #pragma unroll
    for (int i = 0; i < 2; ++i)
        #pragma unroll
        for (int j = 0; j < 4; ++j) acc[i][j] = vzero;

    for (int kt = 0; kt < DMODEL; kt += 64) {
        // B tile (128x64 bf16): pure async DMA, one row per thread (8 x 16B).
        {
            const unsigned short* gB = W + (size_t)(nBase + tid) * DMODEL + kt;
            const unsigned lB = lds_addr(Bs + tid * TSTRIDE);
            #pragma unroll
            for (int e = 0; e < 8; ++e)
                async_load_b128(gB + e * 8, lB + e * 16);
        }
        // A tile (64x64 f32 -> bf16, packed cvt).
        #pragma unroll
        for (int it = 0; it < 8; ++it) {
            const int fi = tid + it * 128;      // 0..1023 float4 slots
            const int r  = fi >> 4;             // row 0..63
            const int c4 = fi & 15;
            float4 xa = *(const float4*)(x + (size_t)(mBase + r) * DMODEL + kt + c4 * 4);
            unsigned* da = (unsigned*)(As + r * TSTRIDE + c4 * 4);
            da[0] = f2bf_pk(xa.x, xa.y);
            da[1] = f2bf_pk(xa.z, xa.w);
        }
        asm volatile("s_wait_asynccnt 0x0" ::: "memory");
        __syncthreads();

        #pragma unroll
        for (int ks = 0; ks < 2; ++ks) {
            v16bf a0 = load_frag_a(As, TSTRIDE, wm + 0,  ks * 32);
            v16bf a1 = load_frag_a(As, TSTRIDE, wm + 16, ks * 32);
            #pragma unroll
            for (int j = 0; j < 4; ++j) {
                v16bf b = load_frag_b(Bs, TSTRIDE, wn + j * 16, ks * 32);
                acc[0][j] = wmma_bf16(a0, b, acc[0][j]);
                acc[1][j] = wmma_bf16(a1, b, acc[1][j]);
            }
        }
        __syncthreads();
    }

    // Epilogue.  A 64-row block never straddles a batch (2048 % 64 == 0).
    const int half8 = (lane < 16) ? 0 : 8;
    const int bidx  = mBase >> 11;            // mBase / SEQ
    const int sBase = mBase & (SEQ - 1);
    #pragma unroll
    for (int i = 0; i < 2; ++i) {
        #pragma unroll
        for (int j = 0; j < 4; ++j) {
            Acc8 a; a.v = acc[i][j];
            const int n = nBase + wn + j * 16 + (lane & 15);
            const float bn = bias[n];
            const int h = n >> 6, d = n & 63;
            unsigned short* op =
                out + (((size_t)bidx * HEADS + h) * SEQ + sBase + wm + i * 16) * DK + d;
            #pragma unroll
            for (int r = 0; r < 8; ++r)
                op[(size_t)(r + half8) * DK] = f2bf((a.f[r] + bn) * outScale);
        }
    }
}

// ---------------------------------------------------------------------------
// Kernel 2: flash attention per (b,h).  Block = 64 query rows, 4 waves,
// each wave owns 16 rows.  K/V tiles double-buffered in LDS, filled with
// global_load_async_to_lds_b128 and pipelined one tile ahead via ASYNCcnt.
// ---------------------------------------------------------------------------
__global__ __launch_bounds__(128) void mha_attention(
    const unsigned short* __restrict__ Q, const unsigned short* __restrict__ K,
    const unsigned short* __restrict__ V, unsigned short* __restrict__ ctx)
{
    const int bh   = blockIdx.y;          // b*HEADS + h
    const int tid  = threadIdx.x;
    const int wave = tid >> 5;
    const int lane = tid & 31;

    const unsigned short* Qh = Q + (size_t)bh * SEQ * DK;
    const unsigned short* Kh = K + (size_t)bh * SEQ * DK;
    const unsigned short* Vh = V + (size_t)bh * SEQ * DK;

    __shared__ unsigned short Kt[2][64 * TSTRIDE];     // [key][d], double buffered
    __shared__ unsigned short Vs[2][64 * TSTRIDE];     // [key][d], double buffered
    __shared__ unsigned short Pt[4][16 * TSTRIDE];     // per-wave P tile [q][key]

    // Preload this wave's Q rows as two A fragments (d = 0..31, 32..63).
    const int qRow0 = blockIdx.x * 64 + wave * 16;
    v16bf qa[2];
    {
        const int m  = qRow0 + (lane & 15);
        const int k0 = (lane < 16) ? 0 : 16;
        const unsigned short* p = Qh + (size_t)m * DK;
        Frag16 f0, f1;
        f0.q[0] = *(const uint4*)(p + k0);
        f0.q[1] = *(const uint4*)(p + k0 + 8);
        f1.q[0] = *(const uint4*)(p + 32 + k0);
        f1.q[1] = *(const uint4*)(p + 32 + k0 + 8);
        qa[0] = f0.v; qa[1] = f1.v;
    }

    float m_i[8], l_i[8];
    v8f od[4];
    const v8f vzero = {0.f,0.f,0.f,0.f,0.f,0.f,0.f,0.f};
    #pragma unroll
    for (int r = 0; r < 8; ++r) { m_i[r] = -1.0e30f; l_i[r] = 0.f; }
    #pragma unroll
    for (int nt = 0; nt < 4; ++nt) od[nt] = vzero;

    const int half8 = (lane < 16) ? 0 : 8;
    unsigned short* PtW = Pt[wave];

    // Each thread DMAs a 64B span of one K row and one V row (8 async ops/tile).
    const int srow = tid >> 1;
    const int sdh  = (tid & 1) * 32;
    auto stage = [&](int kb, int buf) {
        const unsigned short* gK = Kh + (size_t)(kb + srow) * DK + sdh;
        const unsigned short* gV = Vh + (size_t)(kb + srow) * DK + sdh;
        const unsigned lK = lds_addr(&Kt[buf][srow * TSTRIDE + sdh]);
        const unsigned lV = lds_addr(&Vs[buf][srow * TSTRIDE + sdh]);
        #pragma unroll
        for (int e = 0; e < 4; ++e) {
            async_load_b128(gK + e * 8, lK + e * 16);
            async_load_b128(gV + e * 8, lV + e * 16);
        }
    };

    constexpr int NT = SEQ / 64;
    stage(0, 0);

    for (int t = 0; t < NT; ++t) {
        const int buf = t & 1;
        if (t + 1 < NT) {
            stage((t + 1) * 64, buf ^ 1);   // prefetch next tile into other buffer
            // 8 newest async ops are the next tile; <=8 outstanding => tile t done.
            asm volatile("s_wait_asynccnt 0x8" ::: "memory");
        } else {
            asm volatile("s_wait_asynccnt 0x0" ::: "memory");
        }
        __syncthreads();

        const unsigned short* Kb = Kt[buf];
        const unsigned short* Vb = Vs[buf];

        // Scores: S(16q x 64k) = Q . K^T  (K tile row-major = B^T layout).
        v8f sc[4];
        #pragma unroll
        for (int ki = 0; ki < 4; ++ki) {
            v8f c = vzero;
            c = wmma_bf16(qa[0], load_frag_b(Kb, TSTRIDE, ki * 16, 0),  c);
            c = wmma_bf16(qa[1], load_frag_b(Kb, TSTRIDE, ki * 16, 32), c);
            sc[ki] = c;
        }

        // Online softmax (row r lives across the 16-lane half-wave).
        Acc8 sa[4], oa[4];
        #pragma unroll
        for (int ki = 0; ki < 4; ++ki) sa[ki].v = sc[ki];
        #pragma unroll
        for (int nt = 0; nt < 4; ++nt) oa[nt].v = od[nt];

        #pragma unroll
        for (int r = 0; r < 8; ++r) {
            float t0 = fmaxf(fmaxf(sa[0].f[r], sa[1].f[r]),
                             fmaxf(sa[2].f[r], sa[3].f[r]));
            t0 = fmaxf(t0, __shfl_xor(t0, 1, 32));
            t0 = fmaxf(t0, __shfl_xor(t0, 2, 32));
            t0 = fmaxf(t0, __shfl_xor(t0, 4, 32));
            t0 = fmaxf(t0, __shfl_xor(t0, 8, 32));
            const float mnew  = fmaxf(m_i[r], t0);
            const float alpha = __expf(m_i[r] - mnew);
            float psum = 0.f;
            #pragma unroll
            for (int ki = 0; ki < 4; ++ki) {
                const float p = __expf(sa[ki].f[r] - mnew);
                sa[ki].f[r] = p;
                psum += p;
            }
            psum += __shfl_xor(psum, 1, 32);
            psum += __shfl_xor(psum, 2, 32);
            psum += __shfl_xor(psum, 4, 32);
            psum += __shfl_xor(psum, 8, 32);
            l_i[r] = l_i[r] * alpha + psum;
            m_i[r] = mnew;
            #pragma unroll
            for (int nt = 0; nt < 4; ++nt) oa[nt].f[r] *= alpha;
            // Emit P bf16 into this wave's LDS tile (C-layout -> row-major).
            unsigned short* prow = PtW + (r + half8) * TSTRIDE + (lane & 15);
            #pragma unroll
            for (int ki = 0; ki < 4; ++ki) prow[ki * 16] = f2bf(sa[ki].f[r]);
        }
        #pragma unroll
        for (int nt = 0; nt < 4; ++nt) od[nt] = oa[nt].v;

        // PV: O += P . V.  LDS ops are in-order within a wave, so the Pt
        // stores above are visible to these loads without a barrier.
        v16bf pa0 = load_frag_a(PtW, TSTRIDE, 0, 0);
        v16bf pa1 = load_frag_a(PtW, TSTRIDE, 0, 32);
        #pragma unroll
        for (int nt = 0; nt < 4; ++nt) {
            v8f c = od[nt];
            c = wmma_bf16(pa0, load_frag_b_rowmajor(Vb, TSTRIDE, nt * 16, 0),  c);
            c = wmma_bf16(pa1, load_frag_b_rowmajor(Vb, TSTRIDE, nt * 16, 32), c);
            od[nt] = c;
        }
        __syncthreads();   // all waves done with buf before it is refilled
    }

    // Normalize and write ctx (bf16, token-major [B*S, DMODEL]).
    const int b = bh / HEADS, h = bh % HEADS;
    #pragma unroll
    for (int nt = 0; nt < 4; ++nt) {
        Acc8 a; a.v = od[nt];
        const int d = nt * 16 + (lane & 15);
        unsigned short* cp = ctx + ((size_t)b * SEQ + qRow0 + half8) * DMODEL + h * DK + d;
        #pragma unroll
        for (int r = 0; r < 8; ++r)
            cp[(size_t)r * DMODEL] = f2bf(a.f[r] / l_i[r]);
    }
}

// ---------------------------------------------------------------------------
// Kernel 3: output projection.  out = ctx(bf16) @ Wo^T(bf16) + bo  (f32 out)
// Block tile 64(M) x 128(N), 4 waves; each wave = 2x4 WMMA tiles, K-step 64.
// Both operand tiles are pure async DMA into LDS.
// ---------------------------------------------------------------------------
__global__ __launch_bounds__(128) void mha_out_proj(
    const unsigned short* __restrict__ ctx, const unsigned short* __restrict__ Wob,
    const float* __restrict__ bo, float* __restrict__ out)
{
    __shared__ unsigned short As[64 * TSTRIDE];
    __shared__ unsigned short Bs[128 * TSTRIDE];

    const int tid   = threadIdx.x;
    const int wave  = tid >> 5;
    const int lane  = tid & 31;
    const int mBase = blockIdx.x * 64;
    const int nBase = blockIdx.y * 128;
    const int wm = (wave & 1) * 32;
    const int wn = (wave >> 1) * 64;

    v8f acc[2][4];
    const v8f vzero = {0.f,0.f,0.f,0.f,0.f,0.f,0.f,0.f};
    #pragma unroll
    for (int i = 0; i < 2; ++i)
        #pragma unroll
        for (int j = 0; j < 4; ++j) acc[i][j] = vzero;

    const int srow = tid >> 1;
    const int sdh  = (tid & 1) * 32;

    for (int kt = 0; kt < DMODEL; kt += 64) {
        // A tile (64x64): async DMA, 4 x 16B per thread.
        {
            const unsigned short* gA = ctx + (size_t)(mBase + srow) * DMODEL + kt + sdh;
            const unsigned lA = lds_addr(As + srow * TSTRIDE + sdh);
            #pragma unroll
            for (int e = 0; e < 4; ++e)
                async_load_b128(gA + e * 8, lA + e * 16);
        }
        // B tile (128x64): async DMA, one row per thread (8 x 16B).
        {
            const unsigned short* gB = Wob + (size_t)(nBase + tid) * DMODEL + kt;
            const unsigned lB = lds_addr(Bs + tid * TSTRIDE);
            #pragma unroll
            for (int e = 0; e < 8; ++e)
                async_load_b128(gB + e * 8, lB + e * 16);
        }
        asm volatile("s_wait_asynccnt 0x0" ::: "memory");
        __syncthreads();

        #pragma unroll
        for (int ks = 0; ks < 2; ++ks) {
            v16bf a0 = load_frag_a(As, TSTRIDE, wm + 0,  ks * 32);
            v16bf a1 = load_frag_a(As, TSTRIDE, wm + 16, ks * 32);
            #pragma unroll
            for (int j = 0; j < 4; ++j) {
                v16bf b = load_frag_b(Bs, TSTRIDE, wn + j * 16, ks * 32);
                acc[0][j] = wmma_bf16(a0, b, acc[0][j]);
                acc[1][j] = wmma_bf16(a1, b, acc[1][j]);
            }
        }
        __syncthreads();
    }

    const int half8 = (lane < 16) ? 0 : 8;
    #pragma unroll
    for (int i = 0; i < 2; ++i) {
        #pragma unroll
        for (int j = 0; j < 4; ++j) {
            Acc8 a; a.v = acc[i][j];
            const int n = nBase + wn + j * 16 + (lane & 15);
            const float bn = bo[n];
            float* op = out + (size_t)(mBase + wm + i * 16 + half8) * DMODEL + n;
            #pragma unroll
            for (int r = 0; r < 8; ++r)
                op[(size_t)r * DMODEL] = a.f[r] + bn;
        }
    }
}

// ---------------------------------------------------------------------------
extern "C" void kernel_launch(void* const* d_in, const int* in_sizes, int n_in,
                              void* d_out, int out_size, void* d_ws, size_t ws_size,
                              hipStream_t stream) {
    (void)in_sizes; (void)n_in; (void)out_size; (void)ws_size;
    const float* q  = (const float*)d_in[0];
    const float* k  = (const float*)d_in[1];
    const float* v  = (const float*)d_in[2];
    const float* Wq = (const float*)d_in[3];
    const float* bq = (const float*)d_in[4];
    const float* Wk = (const float*)d_in[5];
    const float* bk = (const float*)d_in[6];
    const float* Wv = (const float*)d_in[7];
    const float* bv = (const float*)d_in[8];
    const float* Wo = (const float*)d_in[9];
    const float* bo = (const float*)d_in[10];
    float* out = (float*)d_out;

    const size_t nPerTensor = (size_t)BATCH * HEADS * SEQ * DK;   // 4M bf16
    const size_t nW = (size_t)DMODEL * DMODEL;                    // 1M per weight
    unsigned short* Qw  = (unsigned short*)d_ws;
    unsigned short* Kw  = Qw + nPerTensor;
    unsigned short* Vw  = Kw + nPerTensor;
    unsigned short* Cw  = Vw + nPerTensor;        // ctx: 4096x1024 bf16
    unsigned short* Wqb = Cw + nPerTensor;
    unsigned short* Wkb = Wqb + nW;
    unsigned short* Wvb = Wkb + nW;
    unsigned short* Wob = Wvb + nW;               // total 40 MB of d_ws

    // One-shot weight conversions (bandwidth-bound, ~24 MB total traffic).
    const int cvtGrid = (int)(nW / (256 * 8));
    cvt_f32_bf16<<<cvtGrid, 256, 0, stream>>>(Wq, Wqb, (int)nW);
    cvt_f32_bf16<<<cvtGrid, 256, 0, stream>>>(Wk, Wkb, (int)nW);
    cvt_f32_bf16<<<cvtGrid, 256, 0, stream>>>(Wv, Wvb, (int)nW);
    cvt_f32_bf16<<<cvtGrid, 256, 0, stream>>>(Wo, Wob, (int)nW);

    dim3 gridP(MROWS / 64, DMODEL / 128, 3);
    mha_qkv_proj<<<gridP, 128, 0, stream>>>(q, k, v, Wqb, Wkb, Wvb, bq, bk, bv,
                                            Qw, Kw, Vw);
    dim3 gridA(SEQ / 64, BATCH * HEADS);
    mha_attention<<<gridA, 128, 0, stream>>>(Qw, Kw, Vw, Cw);
    dim3 gridO(MROWS / 64, DMODEL / 128);
    mha_out_proj<<<gridO, 128, 0, stream>>>(Cw, Wob, bo, out);
}
